// Encoder_833223656458
// MI455X (gfx1250) — matrix operand order repaired and checked by reading, not compiled, not run on hardware
//
#include <hip/hip_runtime.h>
#include <hip/hip_bf16.h>
#include <math.h>

// ---------------------------------------------------------------------------
// Types / helpers for CDNA5 (gfx1250) WMMA + TDM, wave32
// ---------------------------------------------------------------------------
typedef __bf16 bf16;
typedef __attribute__((ext_vector_type(16))) __bf16 v16bf;
typedef __attribute__((ext_vector_type(8)))  float  v8f;
typedef __attribute__((ext_vector_type(4))) unsigned int u32x4;
typedef __attribute__((ext_vector_type(8))) int i32x8;
typedef __attribute__((ext_vector_type(4))) int i32x4;

static constexpr int B_  = 4;
static constexpr int S_  = 2048;
static constexpr int DM  = 1024;
static constexpr int NH  = 16;
static constexpr int DH  = 64;
static constexpr int HID = 4096;
static constexpr int TOK = B_ * S_;   // 8192 rows

__device__ __forceinline__ v8f wmma_bf16(v16bf a, v16bf b, v8f c) {
  // D = A(16x32 bf16) * B(32x16 bf16) + C(16x16 f32)
  return __builtin_amdgcn_wmma_f32_16x16x32_bf16(
      /*neg_a=*/false, a, /*neg_b=*/false, b,
      /*c_mod=*/(short)0, c, /*reuse_a=*/false, /*reuse_b=*/false);
}

union FragBF { uint4 u[2]; v16bf v; };
__device__ __forceinline__ v16bf make_frag(uint4 lo, uint4 hi) {
  FragBF f; f.u[0] = lo; f.u[1] = hi; return f.v;
}
__device__ __forceinline__ v8f zero8() {
  v8f z;
  for (int i = 0; i < 8; ++i) z[i] = 0.f;
  return z;
}

// ---------------------------------------------------------------------------
// TDM: issue a 2D tensor_load_to_lds. Tile rows are 32 bf16 (64B); descriptor
// pad feature inserts 16B after every 64B stored -> LDS row stride 40 bf16,
// keeping the WMMA fragment chunks 16B-aligned and bank-spread.
//   lds_off    : byte offset into this workgroup's LDS
//   gaddr      : global address of tile origin
//   tensor_d0  : tensor extent in dim0 (elements)   (OOB clamp)
//   tensor_d1  : tensor extent in dim1 (rows)
//   tile_d1    : rows in tile (tile_d0 fixed = 32)
//   stride0    : elements between consecutive rows in global memory
// ---------------------------------------------------------------------------
__device__ __forceinline__ void tdm_load_tile32(
    unsigned lds_off, const void* gaddr, unsigned tensor_d0,
    unsigned tensor_d1, unsigned tile_d1, unsigned stride0) {
  u32x4 g0;
  g0[0] = 1u;                                   // count=1, user descriptor
  g0[1] = lds_off;                              // lds_addr (bytes)
  unsigned long long ga = (unsigned long long)(size_t)gaddr;
  g0[2] = (unsigned)ga;                         // global_addr[31:0]
  g0[3] = (unsigned)((ga >> 32) & 0x01FFFFFFu)  // global_addr[56:32]
          | 0x80000000u;                        // type = 2 ("image")

  const unsigned tile_d0 = 32u;
  i32x8 g1;
  g1[0] = (int)((1u << 16)      // data_size = 1 -> 2 bytes
              | (1u << 20)      // pad_enable
              | (3u << 22)      // pad_interval: 16 DWORDs (64B)
              | (3u << 25));    // pad_amount:   4 DWORDs (16B)
  g1[1] = (int)((tensor_d0 & 0xFFFFu) << 16);               // dim0[15:0]
  g1[2] = (int)((tensor_d0 >> 16) | ((tensor_d1 & 0xFFFFu) << 16));
  g1[3] = (int)((tensor_d1 >> 16) | (tile_d0 << 16));       // tile_dim0
  g1[4] = (int)tile_d1;                                     // tile_dim1 (dim2=0)
  g1[5] = (int)stride0;                                     // dim0_stride[31:0]
  g1[6] = 0;                                                // stride hi / dim1_stride
  g1[7] = 0;
  i32x4 z4; z4[0] = 0; z4[1] = 0; z4[2] = 0; z4[3] = 0;
  i32x8 z8;
  for (int i = 0; i < 8; ++i) z8[i] = 0;
  __builtin_amdgcn_tensor_load_to_lds(g0, g1, z4, z4, z8, 0);
}

// ---------------------------------------------------------------------------
// fp32 -> bf16 elementwise convert
// ---------------------------------------------------------------------------
__global__ void cvt_bf16(const float* __restrict__ in, bf16* __restrict__ out,
                         size_t n) {
  size_t i = (size_t)blockIdx.x * blockDim.x + threadIdx.x;
  if (i < n) out[i] = (bf16)in[i];
}

// W[K,N] fp32 -> Wt[N,K] bf16 (coalesced writes; one-time cost)
__global__ void transpose_cvt(const float* __restrict__ W, bf16* __restrict__ Wt,
                              int K, int N) {
  size_t i = (size_t)blockIdx.x * blockDim.x + threadIdx.x;
  size_t total = (size_t)K * N;
  if (i >= total) return;
  int n = (int)(i / K);
  int k = (int)(i % K);
  Wt[(size_t)n * K + k] = (bf16)W[(size_t)k * N + n];
}

// ---------------------------------------------------------------------------
// bf16 GEMM with TDM double buffering:
//   C[M,N] = A[M,K](bf16 row-major) @ W  (Wt = [N,K] bf16) + bias
// Block: 256 threads = 8 waves (2 along M x 4 along N), tile 64x256, K-step 32.
// Wave tile 32x64 = 2x4 WMMA tiles (8 v_wmma per K-step).
// Tiles are DMA'd into LDS by the Tensor Data Mover (wave 0 issues; everyone
// synchronizes via s_wait_tensorcnt + workgroup barrier).
// LDS stage layout (row stride 40 bf16 via TDM pad):
//   A tile 64x40  = 5120 B  at  stage*25600
//   B tile 256x40 = 20480 B at  stage*25600 + 5120
// ---------------------------------------------------------------------------
#define GEMM_STAGE_BYTES 25600
#define GEMM_STAGE_ELEMS 12800
#define GEMM_A_ELEMS     2560

template <bool RELU, bool WF32, bool WBF16>
__global__ __launch_bounds__(256) void gemm_bf16(
    const bf16* __restrict__ A, const bf16* __restrict__ Wt,
    const float* __restrict__ bias, float* __restrict__ Cf,
    bf16* __restrict__ Cb, int M, int N, int K) {
  __shared__ __align__(16) bf16 smem[2 * GEMM_STAGE_ELEMS];

  const int t    = threadIdx.x;
  const int lane = t & 31;
  const int hf   = lane >> 4;     // lane half
  const int l16  = lane & 15;
  const int wv   = t >> 5;        // wave 0..7
  const int wm   = wv & 1;        // 2 waves along M (32 rows each)
  const int wn   = wv >> 1;       // 4 waves along N (64 cols each)
  const int m0   = blockIdx.y * 64;
  const int n0   = blockIdx.x * 256;

  v8f acc[2][4];
  for (int mi = 0; mi < 2; ++mi)
    for (int ni = 0; ni < 4; ++ni) acc[mi][ni] = zero8();

  // prologue: stage 0 tiles in flight
  if (wv == 0) {
    tdm_load_tile32(0, A + (size_t)m0 * K, (unsigned)K, (unsigned)M, 64u,
                    (unsigned)K);
    tdm_load_tile32(GEMM_A_ELEMS * 2, Wt + (size_t)n0 * K, (unsigned)K,
                    (unsigned)N, 256u, (unsigned)K);
  }

  int s = 0;
  for (int kt = 0; kt < K; kt += 32, s ^= 1) {
    if (wv == 0) {
      if (kt + 32 < K) {
        const unsigned base = (unsigned)((s ^ 1) * GEMM_STAGE_BYTES);
        tdm_load_tile32(base, A + (size_t)m0 * K + kt + 32, (unsigned)K,
                        (unsigned)M, 64u, (unsigned)K);
        tdm_load_tile32(base + GEMM_A_ELEMS * 2, Wt + (size_t)n0 * K + kt + 32,
                        (unsigned)K, (unsigned)N, 256u, (unsigned)K);
        __builtin_amdgcn_s_wait_tensorcnt(2);  // current stage landed
      } else {
        __builtin_amdgcn_s_wait_tensorcnt(0);
      }
    }
    __syncthreads();   // stage s visible to all waves

    const bf16* Abase = smem + s * GEMM_STAGE_ELEMS;
    const bf16* Bbase = Abase + GEMM_A_ELEMS;

    // A-fragment: lane row = l16; chunks at cols hf*8 and 16+hf*8 (stride 40)
    v16bf af[2];
    for (int mi = 0; mi < 2; ++mi) {
      const bf16* p = Abase + (wm * 32 + mi * 16 + l16) * 40;
      af[mi] = make_frag(*(const uint4*)(p + hf * 8),
                         *(const uint4*)(p + 16 + hf * 8));
    }
    // B-fragment: lane col = l16; chunks at K hf*8 and 16+hf*8
    v16bf bfr[4];
    for (int ni = 0; ni < 4; ++ni) {
      const bf16* p = Bbase + (wn * 64 + ni * 16 + l16) * 40;
      bfr[ni] = make_frag(*(const uint4*)(p + hf * 8), *(const uint4*)(p + 16 + hf * 8));
    }
    for (int mi = 0; mi < 2; ++mi)
      for (int ni = 0; ni < 4; ++ni)
        acc[mi][ni] = wmma_bf16(af[mi], bfr[ni], acc[mi][ni]);

    __syncthreads();   // all reads of stage s done before TDM refills it
  }

  // Epilogue. C layout: VGPR r -> row r (lanes 0-15) / row 8+r (lanes 16-31)
  for (int mi = 0; mi < 2; ++mi)
    for (int ni = 0; ni < 4; ++ni) {
      const int row0 = m0 + wm * 32 + mi * 16 + 8 * hf;
      const int col  = n0 + wn * 64 + ni * 16 + l16;
      const float bv = bias ? bias[col] : 0.f;
      for (int r = 0; r < 8; ++r) {
        float v = acc[mi][ni][r] + bv;
        if (RELU) v = fmaxf(v, 0.f);
        const size_t idx = (size_t)(row0 + r) * N + col;
        if (WF32)  Cf[idx] = v;
        if (WBF16) Cb[idx] = (bf16)v;
      }
    }
}

// ---------------------------------------------------------------------------
// Flash attention: per (b,h), 4 waves x 16 query rows per block.
// Q,K,V,O are bf16 [TOK, DM]; head h occupies columns [h*64, h*64+64).
// ---------------------------------------------------------------------------
__global__ __launch_bounds__(128) void attention(
    const bf16* __restrict__ Qb, const bf16* __restrict__ Kb,
    const bf16* __restrict__ Vb, bf16* __restrict__ Ob) {
  __shared__ __align__(16) bf16 Kl[32][72];      // 32 keys x 64 d (pad 8)
  __shared__ __align__(16) bf16 Vt[64][40];      // transposed: [d][key]
  __shared__ __align__(16) bf16 Pl[4][16][40];   // per-wave P staging

  const int t    = threadIdx.x;
  const int lane = t & 31;
  const int hf   = lane >> 4;
  const int l16  = lane & 15;
  const int wv   = t >> 5;
  const int bh   = blockIdx.y;
  const int b    = bh / NH, h = bh % NH;
  const size_t bs = (size_t)b * S_;
  const int q0   = blockIdx.x * 64 + wv * 16;

  // Q fragments for the two 32-wide K(dim) steps over d=64
  v16bf qf[2];
  {
    const bf16* qp = Qb + (bs + q0 + l16) * DM + h * DH;
    for (int kd = 0; kd < 2; ++kd) {
      const bf16* p = qp + kd * 32;
      qf[kd] = make_frag(*(const uint4*)(p + hf * 8),
                         *(const uint4*)(p + 16 + hf * 8));
    }
  }

  v8f acc[4];
  for (int i = 0; i < 4; ++i) acc[i] = zero8();
  float mrun[8], lrun[8];
  for (int r = 0; r < 8; ++r) { mrun[r] = -1e30f; lrun[r] = 0.f; }

  const int kpr = t >> 2, kc = (t & 3) * 16;   // coop-load indices

  for (int kv0 = 0; kv0 < S_; kv0 += 32) {
    __syncthreads();
    {
      const bf16* kp_ = Kb + (bs + kv0 + kpr) * DM + h * DH + kc;
      *(uint4*)&Kl[kpr][kc]     = *(const uint4*)kp_;
      *(uint4*)&Kl[kpr][kc + 8] = *(const uint4*)(kp_ + 8);
      const bf16* vp_ = Vb + (bs + kv0 + kpr) * DM + h * DH + kc;
      union { uint4 u[2]; bf16 e[16]; } vvu;
      vvu.u[0] = *(const uint4*)vp_;
      vvu.u[1] = *(const uint4*)(vp_ + 8);
      for (int j = 0; j < 16; ++j) Vt[kc + j][kpr] = vvu.e[j];
      if (kv0 + 32 < S_) {  // hint next tiles into cache
        __builtin_prefetch(Kb + (bs + kv0 + 32 + kpr) * DM + h * DH + kc, 0, 1);
        __builtin_prefetch(Vb + (bs + kv0 + 32 + kpr) * DM + h * DH + kc, 0, 1);
      }
    }
    __syncthreads();

    // scores for two 16-key tiles: S = Q @ K^T  (K tile acts as B-frag)
    v8f s[2];
    for (int j = 0; j < 2; ++j) {
      v8f sj = zero8();
      for (int ds = 0; ds < 2; ++ds) {
        const bf16* p = &Kl[j * 16 + l16][ds * 32];
        v16bf kf = make_frag(*(const uint4*)(p + hf * 8), *(const uint4*)(p + 16 + hf * 8));
        sj = wmma_bf16(qf[ds], kf, sj);
      }
      for (int r = 0; r < 8; ++r) sj[r] *= 0.125f;   // 1/sqrt(64)
      s[j] = sj;
    }

    // online softmax (row = r + 8*hf; row stats shared by 16 lanes of half)
    float mt[8], al[8], p0[8], p1[8], rs[8];
    for (int r = 0; r < 8; ++r) mt[r] = fmaxf(s[0][r], s[1][r]);
    for (int off = 1; off < 16; off <<= 1)
      for (int r = 0; r < 8; ++r)
        mt[r] = fmaxf(mt[r], __shfl_xor(mt[r], off, 32));
    for (int r = 0; r < 8; ++r) {
      float mn = fmaxf(mrun[r], mt[r]);
      al[r] = __expf(mrun[r] - mn);
      mrun[r] = mn;
      p0[r] = __expf(s[0][r] - mn);
      p1[r] = __expf(s[1][r] - mn);
      rs[r] = p0[r] + p1[r];
    }
    for (int off = 1; off < 16; off <<= 1)
      for (int r = 0; r < 8; ++r) rs[r] += __shfl_xor(rs[r], off, 32);
    for (int r = 0; r < 8; ++r) lrun[r] = lrun[r] * al[r] + rs[r];
    for (int ni = 0; ni < 4; ++ni)
      for (int r = 0; r < 8; ++r) acc[ni][r] *= al[r];

    // stage P through LDS to reshape C-layout -> A-fragment layout
    for (int r = 0; r < 8; ++r) {
      Pl[wv][r + 8 * hf][l16]      = (bf16)p0[r];
      Pl[wv][r + 8 * hf][16 + l16] = (bf16)p1[r];
    }
    __syncthreads();

    v16bf pf;
    {
      const bf16* p = &Pl[wv][l16][0];
      pf = make_frag(*(const uint4*)(p + hf * 8),
                     *(const uint4*)(p + 16 + hf * 8));
    }
    for (int ni = 0; ni < 4; ++ni) {
      const bf16* p = &Vt[ni * 16 + l16][0];
      v16bf vf = make_frag(*(const uint4*)(p + hf * 8), *(const uint4*)(p + 16 + hf * 8));
      acc[ni] = wmma_bf16(pf, vf, acc[ni]);
    }
  }

  float inv[8];
  for (int r = 0; r < 8; ++r) inv[r] = 1.f / lrun[r];
  for (int ni = 0; ni < 4; ++ni)
    for (int r = 0; r < 8; ++r) {
      const size_t row = bs + q0 + r + 8 * hf;
      Ob[row * DM + h * DH + ni * 16 + l16] = (bf16)(acc[ni][r] * inv[r]);
    }
}

// ---------------------------------------------------------------------------
// Y = LayerNorm(X + R) * gamma + beta ; writes fp32 and optional bf16 copy.
// One block (256 thr) per row of 1024.
// ---------------------------------------------------------------------------
__global__ __launch_bounds__(256) void add_layernorm(
    const float* __restrict__ X, const float* __restrict__ R,
    const float* __restrict__ gamma, const float* __restrict__ beta,
    float* __restrict__ Yf, bf16* __restrict__ Yb) {
  __shared__ float red1[256];
  __shared__ float red2[256];
  const int row = blockIdx.x;
  const int t = threadIdx.x;

  const float4 x4 = ((const float4*)(X + (size_t)row * DM))[t];
  const float4 r4 = ((const float4*)(R + (size_t)row * DM))[t];
  float v[4] = {x4.x + r4.x, x4.y + r4.y, x4.z + r4.z, x4.w + r4.w};
  float s1 = v[0] + v[1] + v[2] + v[3];
  float s2 = v[0] * v[0] + v[1] * v[1] + v[2] * v[2] + v[3] * v[3];
  red1[t] = s1; red2[t] = s2;
  __syncthreads();
  for (int o = 128; o > 0; o >>= 1) {
    if (t < o) { red1[t] += red1[t + o]; red2[t] += red2[t + o]; }
    __syncthreads();
  }
  const float mu   = red1[0] * (1.f / DM);
  const float var  = red2[0] * (1.f / DM) - mu * mu;
  const float rstd = rsqrtf(var + 1e-5f);

  const float4 g4 = ((const float4*)gamma)[t];
  const float4 b4 = ((const float4*)beta)[t];
  float o0 = (v[0] - mu) * rstd * g4.x + b4.x;
  float o1 = (v[1] - mu) * rstd * g4.y + b4.y;
  float o2 = (v[2] - mu) * rstd * g4.z + b4.z;
  float o3 = (v[3] - mu) * rstd * g4.w + b4.w;
  ((float4*)(Yf + (size_t)row * DM))[t] = make_float4(o0, o1, o2, o3);
  if (Yb) {
    bf16* yp = Yb + (size_t)row * DM + t * 4;
    yp[0] = (bf16)o0; yp[1] = (bf16)o1; yp[2] = (bf16)o2; yp[3] = (bf16)o3;
  }
}

// ---------------------------------------------------------------------------
// Host orchestration
// ---------------------------------------------------------------------------
extern "C" void kernel_launch(void* const* d_in, const int* in_sizes, int n_in,
                              void* d_out, int out_size, void* d_ws,
                              size_t ws_size, hipStream_t stream) {
  const float* X   = (const float*)d_in[0];
  const float* Wq  = (const float*)d_in[1];
  const float* bq  = (const float*)d_in[2];
  const float* Wk  = (const float*)d_in[3];
  const float* bk  = (const float*)d_in[4];
  const float* Wv  = (const float*)d_in[5];
  const float* bv  = (const float*)d_in[6];
  const float* Wo  = (const float*)d_in[7];
  const float* bo  = (const float*)d_in[8];
  const float* g1  = (const float*)d_in[9];
  const float* be1 = (const float*)d_in[10];
  const float* W1  = (const float*)d_in[11];
  const float* b1  = (const float*)d_in[12];
  const float* W2  = (const float*)d_in[13];
  const float* b2  = (const float*)d_in[14];
  const float* g2  = (const float*)d_in[15];
  const float* be2 = (const float*)d_in[16];
  float* Out = (float*)d_out;

  char* ws = (char*)d_ws;
  size_t off = 0;
  auto alloc = [&](size_t bytes) -> char* {
    char* p = ws + off;
    off += (bytes + 255) & ~(size_t)255;
    return p;
  };
  bf16* Xb   = (bf16*)alloc((size_t)TOK * DM * 2);
  bf16* WqT  = (bf16*)alloc((size_t)DM * DM * 2);
  bf16* WkT  = (bf16*)alloc((size_t)DM * DM * 2);
  bf16* WvT  = (bf16*)alloc((size_t)DM * DM * 2);
  bf16* WoT  = (bf16*)alloc((size_t)DM * DM * 2);
  bf16* W1T  = (bf16*)alloc((size_t)DM * HID * 2);
  bf16* W2T  = (bf16*)alloc((size_t)HID * DM * 2);
  bf16* Qb   = (bf16*)alloc((size_t)TOK * DM * 2);
  bf16* Kb   = (bf16*)alloc((size_t)TOK * DM * 2);
  bf16* Vb   = (bf16*)alloc((size_t)TOK * DM * 2);
  bf16* Ctxb = (bf16*)alloc((size_t)TOK * DM * 2);
  float* Attn = (float*)alloc((size_t)TOK * DM * 4);
  float* Yf   = (float*)alloc((size_t)TOK * DM * 4);
  bf16*  Yb   = (bf16*)alloc((size_t)TOK * DM * 2);
  bf16*  Hb   = (bf16*)alloc((size_t)TOK * HID * 2);
  float* Ffn  = (float*)alloc((size_t)TOK * DM * 4);

  // 1) convert activations / weights to bf16 (weights transposed to [N,K])
  {
    size_t nX = (size_t)TOK * DM;
    cvt_bf16<<<dim3((unsigned)((nX + 255) / 256)), dim3(256), 0, stream>>>(X, Xb, nX);
    size_t nW = (size_t)DM * DM;
    unsigned gW = (unsigned)((nW + 255) / 256);
    transpose_cvt<<<gW, 256, 0, stream>>>(Wq, WqT, DM, DM);
    transpose_cvt<<<gW, 256, 0, stream>>>(Wk, WkT, DM, DM);
    transpose_cvt<<<gW, 256, 0, stream>>>(Wv, WvT, DM, DM);
    transpose_cvt<<<gW, 256, 0, stream>>>(Wo, WoT, DM, DM);
    size_t nF = (size_t)DM * HID;
    unsigned gF = (unsigned)((nF + 255) / 256);
    transpose_cvt<<<gF, 256, 0, stream>>>(W1, W1T, DM, HID);   // -> [HID, DM]
    transpose_cvt<<<gF, 256, 0, stream>>>(W2, W2T, HID, DM);   // -> [DM, HID]
  }

  // 2) Q/K/V projections (bf16 out)
  {
    dim3 g(DM / 256, TOK / 64);
    gemm_bf16<false, false, true><<<g, 256, 0, stream>>>(Xb, WqT, bq, nullptr, Qb, TOK, DM, DM);
    gemm_bf16<false, false, true><<<g, 256, 0, stream>>>(Xb, WkT, bk, nullptr, Kb, TOK, DM, DM);
    gemm_bf16<false, false, true><<<g, 256, 0, stream>>>(Xb, WvT, bv, nullptr, Vb, TOK, DM, DM);
  }

  // 3) attention
  {
    dim3 g(S_ / 64, B_ * NH);
    attention<<<g, 128, 0, stream>>>(Qb, Kb, Vb, Ctxb);
  }

  // 4) output projection (f32 out) + AddNorm1
  {
    dim3 g(DM / 256, TOK / 64);
    gemm_bf16<false, true, false><<<g, 256, 0, stream>>>(Ctxb, WoT, bo, Attn, nullptr, TOK, DM, DM);
    add_layernorm<<<TOK, 256, 0, stream>>>(X, Attn, g1, be1, Yf, Yb);
  }

  // 5) FFN + AddNorm2
  {
    dim3 g1d(HID / 256, TOK / 64);
    gemm_bf16<true, false, true><<<g1d, 256, 0, stream>>>(Yb, W1T, b1, nullptr, Hb, TOK, HID, DM);
    dim3 g2d(DM / 256, TOK / 64);
    gemm_bf16<false, true, false><<<g2d, 256, 0, stream>>>(Hb, W2T, b2, Ffn, nullptr, TOK, DM, HID);
    add_layernorm<<<TOK, 256, 0, stream>>>(Yf, Ffn, g2, be2, Out, nullptr);
  }
}
